// MSEmbeddingNormNet_47356309405763
// MI455X (gfx1250) — matrix operand order repaired and checked by reading, not compiled, not run on hardware
//
#include <hip/hip_runtime.h>
#include <hip/hip_bf16.h>
#include <stddef.h>
#include <stdint.h>

typedef __bf16 bf16_t;
typedef __attribute__((ext_vector_type(16))) __bf16 v16bf;
typedef __attribute__((ext_vector_type(8)))  __bf16 v8bf;
typedef __attribute__((ext_vector_type(8)))  float  v8f;
typedef __attribute__((ext_vector_type(4)))  unsigned int v4u;
typedef __attribute__((ext_vector_type(8)))  int v8i;
typedef __attribute__((ext_vector_type(4)))  int v4i;

#define BB 64
#define TT 512
#define FF 32
#define DD 64
#define HH 512
#define G4 2048                     // 4*H
#define SLAB_BYTES (128 * 32 * 16 * 2)   // one kt-slab of packed W_hh: 128 KB
#define SLAB_ELEMS (128 * 32 * 16)       // in bf16 elements

// ---------------------------------------------------------------------------
// helpers
// ---------------------------------------------------------------------------
static __device__ __forceinline__ v16bf load_a_frag(const bf16_t* p) {
  // A-operand (16-bit, 16x32): per lane two contiguous 8-element (16B) chunks:
  //   e0..7  -> K = base .. base+7 ;  e8..15 -> K = base+8 .. base+15
  union { v16bf v; v8bf h[2]; } u;
  u.h[0] = *(const v8bf*)(p);
  u.h[1] = *(const v8bf*)(p + 8);
  return u.v;
}

static __device__ __forceinline__ float sigmoidf_(float x) {
  return 1.0f / (1.0f + __expf(-x));
}

// sum across the 16 lanes of each wave32 half (rows are split by lane half)
static __device__ __forceinline__ float half16_sum(float v) {
  v += __shfl_xor(v, 1, 32);
  v += __shfl_xor(v, 2, 32);
  v += __shfl_xor(v, 4, 32);
  v += __shfl_xor(v, 8, 32);
  return v;
}

// Issue a Tensor Data Mover DMA of one contiguous 128KB slab: global -> LDS.
// Descriptor per CDNA5 ISA sec.8: 1D tile, data_size=8B,
// tile_dim0 = 16384 elements, D# type=2, count=1. Wave-uniform arguments.
static __device__ __forceinline__ void tdm_load_slab(const bf16_t* gsrc, unsigned int lds_byte_addr) {
  const uint64_t ga = (uint64_t)(uintptr_t)gsrc;
  const unsigned int elems = SLAB_BYTES / 8;   // 16384 x 8-byte elements
  v4u g0;
  g0[0] = 1u;                                          // count=1, is_restore=0
  g0[1] = lds_byte_addr;                               // lds_addr @ bits 63:32
  g0[2] = (unsigned int)ga;                            // global_addr[31:0] @ 95:64
  g0[3] = (unsigned int)((ga >> 32) & 0x01ffffffu) | (2u << 30); // addr[56:32] + type=2
  v8i g1;
  g1[0] = (int)(3u << 16);                             // data_size=3 (8B), wg_mask=0
  g1[1] = (int)((elems & 0xffffu) << 16);              // tensor_dim0[15:0] @ 63:48
  g1[2] = (int)(((elems >> 16) & 0xffffu) | (1u << 16)); // tensor_dim0[31:16] @ 79:64, tensor_dim1=1 @ 95:80
  g1[3] = (int)((elems & 0xffffu) << 16);              // tile_dim0 @ 127:112
  g1[4] = 0;                                           // tile_dim1=0 (unused -> 1D), tile_dim2=0
  g1[5] = (int)elems;                                  // tensor_dim0_stride (lo)
  g1[6] = 0;
  g1[7] = 0;
  v4i g2 = {0, 0, 0, 0};
  v4i g3 = {0, 0, 0, 0};
#if __clang_major__ >= 23
  v8i g1b = {0, 0, 0, 0, 0, 0, 0, 0};
  __builtin_amdgcn_tensor_load_to_lds(g0, g1, g2, g3, g1b, 0);
#else
  __builtin_amdgcn_tensor_load_to_lds(g0, g1, g2, g3, 0);
#endif
}

// ---------------------------------------------------------------------------
// 1) Peak embedding: two 1->32->32 MLPs, relu, length mask -> bf16 [B][T][D]
// ---------------------------------------------------------------------------
__global__ void embed_kernel(const float* __restrict__ peaks,
                             const int*   __restrict__ plen,
                             const float* __restrict__ w_mz1,  const float* __restrict__ b_mz1,
                             const float* __restrict__ w_mz2,  const float* __restrict__ b_mz2,
                             const float* __restrict__ w_int1, const float* __restrict__ b_int1,
                             const float* __restrict__ w_int2, const float* __restrict__ b_int2,
                             bf16_t* __restrict__ xbf) {
  int idx = blockIdx.x * blockDim.x + threadIdx.x;   // b*T + t
  if (idx >= BB * TT) return;
  int b = idx / TT, t = idx % TT;
  bool masked = t >= (TT - plen[b]);
  float mz = peaks[(size_t)idx * 2 + 0];
  float it = peaks[(size_t)idx * 2 + 1];

  float h1[32];
  #pragma unroll
  for (int i = 0; i < 32; ++i) { float v = mz * w_mz1[i] + b_mz1[i]; h1[i] = v > 0.f ? v : 0.f; }
  for (int f = 0; f < 32; ++f) {
    float s = b_mz2[f];
    #pragma unroll
    for (int i = 0; i < 32; ++i) s += w_mz2[f * 32 + i] * h1[i];
    s = s > 0.f ? s : 0.f;
    xbf[(size_t)idx * DD + f] = (bf16_t)(masked ? 0.f : s);
  }
  #pragma unroll
  for (int i = 0; i < 32; ++i) { float v = it * w_int1[i] + b_int1[i]; h1[i] = v > 0.f ? v : 0.f; }
  for (int f = 0; f < 32; ++f) {
    float s = b_int2[f];
    #pragma unroll
    for (int i = 0; i < 32; ++i) s += w_int2[f * 32 + i] * h1[i];
    s = s > 0.f ? s : 0.f;
    xbf[(size_t)idx * DD + 32 + f] = (bf16_t)(masked ? 0.f : s);
  }
}

// ---------------------------------------------------------------------------
// 2) Weight packing into wave32 WMMA B-operand layout (bf16):
//    element (lane, e) of tile (n_tile, k_tile):  N = n_tile*16 + (lane&15),
//    K = k_tile*32 + e + 16*(lane>=16).  B[k][n] = W[n][k] (x @ W^T).
//    W_hh is packed slab-major ([dir][kt][nt][lane][e]) so each kt-slab is one
//    contiguous 128KB block for the TDM.
// ---------------------------------------------------------------------------
__global__ void pack_whh_kernel(const float* __restrict__ w, bf16_t* __restrict__ p) {
  int i = blockIdx.x * blockDim.x + threadIdx.x;     // [dir][kt=16][nt=128][lane=32][e=16]
  if (i >= 2 * 16 * 128 * 32 * 16) return;
  int e = i & 15, lane = (i >> 4) & 31, nt = (i >> 9) & 127, kt = (i >> 16) & 15, dir = (i >> 20) & 1;
  int n = nt * 16 + (lane & 15);
  int k = kt * 32 + e + ((lane >> 4) & 1) * 16;
  p[i] = (bf16_t)w[((size_t)dir * G4 + n) * HH + k];
}

__global__ void pack_wih_kernel(const float* __restrict__ w, bf16_t* __restrict__ p) {
  int i = blockIdx.x * blockDim.x + threadIdx.x;     // [dir][nt=128][kt=2][lane=32][e=16]
  if (i >= 2 * 128 * 2 * 32 * 16) return;
  int e = i & 15, lane = (i >> 4) & 31, kt = (i >> 9) & 1, nt = (i >> 10) & 127, dir = (i >> 17) & 1;
  int n = nt * 16 + (lane & 15);
  int k = kt * 32 + e + ((lane >> 4) & 1) * 16;
  p[i] = (bf16_t)w[((size_t)dir * G4 + n) * DD + k];
}

__global__ void pack_wfuse_kernel(const float* __restrict__ w, bf16_t* __restrict__ p) {
  int i = blockIdx.x * blockDim.x + threadIdx.x;     // [nt=32][kt=32][lane=32][e=16]
  if (i >= 32 * 32 * 32 * 16) return;
  int e = i & 15, lane = (i >> 4) & 31, kt = (i >> 9) & 31, nt = (i >> 14) & 31;
  int n = nt * 16 + (lane & 15);
  int k = kt * 32 + e + ((lane >> 4) & 1) * 16;
  p[i] = (bf16_t)w[(size_t)n * (2 * HH) + k];
}

// ---------------------------------------------------------------------------
// 3) state init: h0 -> bf16 h_state, c0 -> f32 c_state   (layout [dir][B][H])
// ---------------------------------------------------------------------------
__global__ void init_state_kernel(const float* __restrict__ h0, const float* __restrict__ c0,
                                  bf16_t* __restrict__ h_state, float* __restrict__ c_state) {
  int i = blockIdx.x * blockDim.x + threadIdx.x;
  if (i >= 2 * BB * HH) return;
  h_state[i] = (bf16_t)h0[i];
  c_state[i] = c0[i];
}

// ---------------------------------------------------------------------------
// 4) one LSTM timestep, both directions.
//    grid = (m_tile=4, dir=2), block = 512 threads (16 waves).
//    Wave w owns n_tiles {w + 16k}; gates live in WMMA accumulator registers.
//    W_hh kt-slabs are streamed global->LDS by the Tensor Data Mover (wave 0),
//    double buffered (2 x 128KB of the 320KB WGP LDS), overlapping compute.
// ---------------------------------------------------------------------------
__global__ void __launch_bounds__(512) lstm_step_kernel(
    const bf16_t* __restrict__ xbf,
    const bf16_t* __restrict__ wih_p,
    const bf16_t* __restrict__ whh_p,
    const float* __restrict__ lni_s, const float* __restrict__ lni_b,
    const float* __restrict__ lnh_s, const float* __restrict__ lnh_b,
    const float* __restrict__ lnc_s, const float* __restrict__ lnc_b,
    bf16_t* __restrict__ h_state, float* __restrict__ c_state, int t) {
  extern __shared__ char dynsmem[];                 // 2 x 128KB W_hh slab buffers
  bf16_t* slab = (bf16_t*)dynsmem;
  __shared__ float red_sx[16 * 16];                 // [row][wave]
  __shared__ float red_qx[16 * 16];
  __shared__ float red_sh[16 * 16];
  __shared__ float red_qh[16 * 16];
  __shared__ float st[96];   // mu_x[16] mu_h[16] rs_x[16] rs_h[16] mu_c[16] rs_c[16]

  const int m_tile = blockIdx.x;             // 0..3
  const int dir    = blockIdx.y;             // 0..1
  const int tid    = threadIdx.x;
  const int wave   = tid >> 5;               // 0..15
  const int lane   = tid & 31;
  const int lhalf  = (lane >> 4) & 1;
  const int l15    = lane & 15;
  const int t_x    = dir ? (TT - 1 - t) : t;

  const bf16_t* whh_dir = whh_p + (size_t)dir * 16 * SLAB_ELEMS;

  // kick off TDM for kt=0 slab; it runs while we do the x-GEMM.
  if (wave == 0) {
    tdm_load_slab(whh_dir, (unsigned int)(uintptr_t)slab);
  }

  const v8f vzero = {0.f, 0.f, 0.f, 0.f, 0.f, 0.f, 0.f, 0.f};
  v8f accx[8], acch[8];
  #pragma unroll
  for (int nt = 0; nt < 8; ++nt) { accx[nt] = vzero; acch[nt] = vzero; }

#define B_IH(kt, nt) (*(const v16bf*)(wih_p + ((((size_t)dir * 128 + (wave + 16 * (nt))) * 2 + (kt)) * 32 + lane) * 16))
#define B_LDS(buf, nt) (*(const v16bf*)(slab + (size_t)(buf) * SLAB_ELEMS + (((wave + 16 * (nt)) * 32) + lane) * 16))

  // ---- gates_x = x_t @ w_ih^T   (M=16, N=2048, K=64), from global/L2 ----
  const bf16_t* xrow = xbf + ((size_t)(m_tile * 16 + l15) * TT + t_x) * DD;
  #pragma unroll
  for (int kt = 0; kt < 2; ++kt) {
    v16bf a = load_a_frag(xrow + kt * 32 + lhalf * 16);
    v16bf b0 = B_IH(kt, 0), b1 = B_IH(kt, 1), b2 = B_IH(kt, 2), b3 = B_IH(kt, 3);
    accx[0] = __builtin_amdgcn_wmma_f32_16x16x32_bf16(false, a, false, b0, (short)0, accx[0], false, false);
    accx[1] = __builtin_amdgcn_wmma_f32_16x16x32_bf16(false, a, false, b1, (short)0, accx[1], false, false);
    accx[2] = __builtin_amdgcn_wmma_f32_16x16x32_bf16(false, a, false, b2, (short)0, accx[2], false, false);
    accx[3] = __builtin_amdgcn_wmma_f32_16x16x32_bf16(false, a, false, b3, (short)0, accx[3], false, false);
    v16bf b4 = B_IH(kt, 4), b5 = B_IH(kt, 5), b6 = B_IH(kt, 6), b7 = B_IH(kt, 7);
    accx[4] = __builtin_amdgcn_wmma_f32_16x16x32_bf16(false, a, false, b4, (short)0, accx[4], false, false);
    accx[5] = __builtin_amdgcn_wmma_f32_16x16x32_bf16(false, a, false, b5, (short)0, accx[5], false, false);
    accx[6] = __builtin_amdgcn_wmma_f32_16x16x32_bf16(false, a, false, b6, (short)0, accx[6], false, false);
    accx[7] = __builtin_amdgcn_wmma_f32_16x16x32_bf16(false, a, false, b7, (short)0, accx[7], false, false);
  }

  // wait for slab 0, make it visible to all waves
  if (wave == 0) __builtin_amdgcn_s_wait_tensorcnt(0);
  __syncthreads();

  // ---- gates_h = h @ w_hh^T   (M=16, N=2048, K=512), B from LDS slabs ----
  const bf16_t* hrow = h_state + ((size_t)dir * BB + m_tile * 16 + l15) * HH;
  #pragma unroll
  for (int kt = 0; kt < 16; ++kt) {
    const int buf = kt & 1;
    // stream next slab into the other buffer while we compute this one
    if (wave == 0 && kt < 15) {
      tdm_load_slab(whh_dir + (size_t)(kt + 1) * SLAB_ELEMS,
                    (unsigned int)(uintptr_t)(slab + (size_t)(1 - buf) * SLAB_ELEMS));
    }
    v16bf a = load_a_frag(hrow + kt * 32 + lhalf * 16);
    v16bf b0 = B_LDS(buf, 0), b1 = B_LDS(buf, 1), b2 = B_LDS(buf, 2), b3 = B_LDS(buf, 3);
    acch[0] = __builtin_amdgcn_wmma_f32_16x16x32_bf16(false, a, false, b0, (short)0, acch[0], false, false);
    acch[1] = __builtin_amdgcn_wmma_f32_16x16x32_bf16(false, a, false, b1, (short)0, acch[1], false, false);
    acch[2] = __builtin_amdgcn_wmma_f32_16x16x32_bf16(false, a, false, b2, (short)0, acch[2], false, false);
    acch[3] = __builtin_amdgcn_wmma_f32_16x16x32_bf16(false, a, false, b3, (short)0, acch[3], false, false);
    v16bf b4 = B_LDS(buf, 4), b5 = B_LDS(buf, 5), b6 = B_LDS(buf, 6), b7 = B_LDS(buf, 7);
    acch[4] = __builtin_amdgcn_wmma_f32_16x16x32_bf16(false, a, false, b4, (short)0, acch[4], false, false);
    acch[5] = __builtin_amdgcn_wmma_f32_16x16x32_bf16(false, a, false, b5, (short)0, acch[5], false, false);
    acch[6] = __builtin_amdgcn_wmma_f32_16x16x32_bf16(false, a, false, b6, (short)0, acch[6], false, false);
    acch[7] = __builtin_amdgcn_wmma_f32_16x16x32_bf16(false, a, false, b7, (short)0, acch[7], false, false);
    if (kt < 15) {
      if (wave == 0) __builtin_amdgcn_s_wait_tensorcnt(0);
      __syncthreads();
    }
  }
#undef B_IH
#undef B_LDS

  // ---- LN row statistics over 2048 (gx and gh), straight from registers ----
  #pragma unroll
  for (int r = 0; r < 8; ++r) {
    float sx = 0.f, qx = 0.f, sh = 0.f, qh = 0.f;
    #pragma unroll
    for (int nt = 0; nt < 8; ++nt) {
      float vx = accx[nt][r]; sx += vx; qx += vx * vx;
      float vh = acch[nt][r]; sh += vh; qh += vh * vh;
    }
    sx = half16_sum(sx); qx = half16_sum(qx);
    sh = half16_sum(sh); qh = half16_sum(qh);
    if (l15 == 0) {
      const int row = r + 8 * lhalf;
      red_sx[row * 16 + wave] = sx; red_qx[row * 16 + wave] = qx;
      red_sh[row * 16 + wave] = sh; red_qh[row * 16 + wave] = qh;
    }
  }
  __syncthreads();
  if (tid < 32) {
    const int r2 = tid & 15;
    const bool isH = tid >= 16;
    const float* rs = isH ? red_sh : red_sx;
    const float* rq = isH ? red_qh : red_qx;
    float s = 0.f, q = 0.f;
    for (int i = 0; i < 16; ++i) { s += rs[r2 * 16 + i]; q += rq[r2 * 16 + i]; }
    const float mu  = s * (1.0f / 2048.0f);
    const float var = q * (1.0f / 2048.0f) - mu * mu;
    st[(isH ? 16 : 0) + r2]      = mu;
    st[32 + (isH ? 16 : 0) + r2] = rsqrtf(var + 1e-5f);
  }
  __syncthreads();

  // ---- gates + cell candidate, entirely in registers ----
  const size_t cb_rows = (size_t)dir * BB + m_tile * 16;
  float cpre[2][8], oraw[2][8];
  #pragma unroll
  for (int c = 0; c < 2; ++c) {
    const int j  = (wave + 16 * c) * 16 + l15;   // owned gate column
    const float si_i = lni_s[dir * G4 + j],            bi_i = lni_b[dir * G4 + j];
    const float si_f = lni_s[dir * G4 + HH + j],       bi_f = lni_b[dir * G4 + HH + j];
    const float si_g = lni_s[dir * G4 + 2 * HH + j],   bi_g = lni_b[dir * G4 + 2 * HH + j];
    const float si_o = lni_s[dir * G4 + 3 * HH + j],   bi_o = lni_b[dir * G4 + 3 * HH + j];
    const float sh_i = lnh_s[dir * G4 + j],            bh_i = lnh_b[dir * G4 + j];
    const float sh_f = lnh_s[dir * G4 + HH + j],       bh_f = lnh_b[dir * G4 + HH + j];
    const float sh_g = lnh_s[dir * G4 + 2 * HH + j],   bh_g = lnh_b[dir * G4 + 2 * HH + j];
    const float sh_o = lnh_s[dir * G4 + 3 * HH + j],   bh_o = lnh_b[dir * G4 + 3 * HH + j];
    #pragma unroll
    for (int r = 0; r < 8; ++r) {
      const int row = r + 8 * lhalf;
      const float mux = st[row], muh = st[16 + row];
      const float rsx = st[32 + row], rsh = st[48 + row];
      const float ig = (accx[0 + c][r] - mux) * rsx * si_i + bi_i + (acch[0 + c][r] - muh) * rsh * sh_i + bh_i;
      const float fg = (accx[2 + c][r] - mux) * rsx * si_f + bi_f + (acch[2 + c][r] - muh) * rsh * sh_f + bh_f;
      const float gg = (accx[4 + c][r] - mux) * rsx * si_g + bi_g + (acch[4 + c][r] - muh) * rsh * sh_g + bh_g;
      const float og = (accx[6 + c][r] - mux) * rsx * si_o + bi_o + (acch[6 + c][r] - muh) * rsh * sh_o + bh_o;
      const float c_old = c_state[(cb_rows + row) * HH + j];
      cpre[c][r] = sigmoidf_(fg) * c_old + sigmoidf_(ig) * tanhf(gg);
      oraw[c][r] = og;
    }
  }

  // ---- LN_c statistics over 512 per row ----
  #pragma unroll
  for (int r = 0; r < 8; ++r) {
    float s = cpre[0][r] + cpre[1][r];
    float q = cpre[0][r] * cpre[0][r] + cpre[1][r] * cpre[1][r];
    s = half16_sum(s); q = half16_sum(q);
    if (l15 == 0) {
      const int row = r + 8 * lhalf;
      red_sx[row * 16 + wave] = s;
      red_qx[row * 16 + wave] = q;
    }
  }
  __syncthreads();
  if (tid < 16) {
    float s = 0.f, q = 0.f;
    for (int i = 0; i < 16; ++i) { s += red_sx[tid * 16 + i]; q += red_qx[tid * 16 + i]; }
    const float mu  = s * (1.0f / 512.0f);
    const float var = q * (1.0f / 512.0f) - mu * mu;
    st[64 + tid] = mu;
    st[80 + tid] = rsqrtf(var + 1e-5f);
  }
  __syncthreads();

  // ---- c2 = LN_c(cpre), h2 = sigmoid(o)*tanh(c2) ----
  #pragma unroll
  for (int c = 0; c < 2; ++c) {
    const int j = (wave + 16 * c) * 16 + l15;
    const float cs = lnc_s[dir * HH + j], cb = lnc_b[dir * HH + j];
    #pragma unroll
    for (int r = 0; r < 8; ++r) {
      const int row = r + 8 * lhalf;
      const float c2 = (cpre[c][r] - st[64 + row]) * st[80 + row] * cs + cb;
      const float h2 = sigmoidf_(oraw[c][r]) * tanhf(c2);
      c_state[(cb_rows + row) * HH + j] = c2;
      h_state[(cb_rows + row) * HH + j] = (bf16_t)h2;
    }
  }
}

// ---------------------------------------------------------------------------
// 5) concat(c_fwd, c_bwd) -> bf16 feature matrix [B][2H]
// ---------------------------------------------------------------------------
__global__ void feat_convert_kernel(const float* __restrict__ c_state, bf16_t* __restrict__ featbf) {
  int i = blockIdx.x * blockDim.x + threadIdx.x;
  if (i >= BB * 2 * HH) return;
  int b = i >> 10, kk = i & 1023;
  int dir = kk >> 9, j = kk & 511;
  featbf[i] = (bf16_t)c_state[((size_t)dir * BB + b) * HH + j];
}

// ---------------------------------------------------------------------------
// 6) fuse GEMM: out = relu(feat @ w_fuse^T + b_fuse)   (64x512, K=1024)
//    one wave per 16x16 tile.
// ---------------------------------------------------------------------------
__global__ void fuse_kernel(const bf16_t* __restrict__ featbf, const bf16_t* __restrict__ wf_p,
                            const float* __restrict__ b_fuse, float* __restrict__ out) {
  const int tile   = blockIdx.x;    // 0..127
  const int m_tile = tile >> 5;     // 0..3
  const int n_tile = tile & 31;     // 0..31
  const int lane   = threadIdx.x;   // 0..31
  const int lhalf  = (lane >> 4) & 1;
  const int l15    = lane & 15;

  v8f acc = {0.f, 0.f, 0.f, 0.f, 0.f, 0.f, 0.f, 0.f};
  const bf16_t* arow = featbf + (size_t)(m_tile * 16 + l15) * (2 * HH);
  #pragma unroll 4
  for (int kt = 0; kt < 32; ++kt) {
    v16bf a = load_a_frag(arow + kt * 32 + lhalf * 16);
    v16bf b = *(const v16bf*)(wf_p + (((size_t)n_tile * 32 + kt) * 32 + lane) * 16);
    acc = __builtin_amdgcn_wmma_f32_16x16x32_bf16(false, a, false, b, (short)0, acc, false, false);
  }
  const int col = n_tile * 16 + l15;
  #pragma unroll
  for (int r = 0; r < 8; ++r) {
    const int rowg = m_tile * 16 + r + lhalf * 8;
    float v = acc[r] + b_fuse[col];
    out[(size_t)rowg * HH + col] = v > 0.f ? v : 0.f;
  }
}

// ---------------------------------------------------------------------------
// launch
// ---------------------------------------------------------------------------
extern "C" void kernel_launch(void* const* d_in, const int* in_sizes, int n_in,
                              void* d_out, int out_size, void* d_ws, size_t ws_size,
                              hipStream_t stream) {
  const float* peaks  = (const float*)d_in[0];
  const int*   plen   = (const int*)  d_in[1];
  const float* w_mz1  = (const float*)d_in[2];
  const float* b_mz1  = (const float*)d_in[3];
  const float* w_mz2  = (const float*)d_in[4];
  const float* b_mz2  = (const float*)d_in[5];
  const float* w_int1 = (const float*)d_in[6];
  const float* b_int1 = (const float*)d_in[7];
  const float* w_int2 = (const float*)d_in[8];
  const float* b_int2 = (const float*)d_in[9];
  const float* w_ih   = (const float*)d_in[10];
  const float* w_hh   = (const float*)d_in[11];
  const float* lni_s  = (const float*)d_in[12];
  const float* lni_b  = (const float*)d_in[13];
  const float* lnh_s  = (const float*)d_in[14];
  const float* lnh_b  = (const float*)d_in[15];
  const float* lnc_s  = (const float*)d_in[16];
  const float* lnc_b  = (const float*)d_in[17];
  const float* w_fuse = (const float*)d_in[18];
  const float* b_fuse = (const float*)d_in[19];
  const float* h0     = (const float*)d_in[20];
  const float* c0     = (const float*)d_in[21];
  float* out = (float*)d_out;

  char* ws = (char*)d_ws;
  size_t off = 0;
  auto carve = [&](size_t bytes) -> void* {
    void* p = ws + off;
    off = (off + bytes + 255) & ~(size_t)255;
    return p;
  };
  bf16_t* xbf     = (bf16_t*)carve((size_t)BB * TT * DD * 2);           // 4 MB
  bf16_t* whh_p   = (bf16_t*)carve((size_t)2 * 16 * SLAB_ELEMS * 2);    // 4 MB, slab-major
  bf16_t* wih_p   = (bf16_t*)carve((size_t)2 * 128 * 2 * 32 * 16 * 2);  // 512 KB
  bf16_t* wf_p    = (bf16_t*)carve((size_t)32 * 32 * 32 * 16 * 2);      // 1 MB
  bf16_t* h_state = (bf16_t*)carve((size_t)2 * BB * HH * 2);
  float*  c_state = (float*) carve((size_t)2 * BB * HH * 4);
  bf16_t* featbf  = (bf16_t*)carve((size_t)BB * 2 * HH * 2);

  const int step_smem = 2 * SLAB_BYTES;   // 256KB double-buffered W_hh slabs
  (void)hipFuncSetAttribute(reinterpret_cast<const void*>(&lstm_step_kernel),
                            hipFuncAttributeMaxDynamicSharedMemorySize, step_smem);

  embed_kernel<<<(BB * TT + 255) / 256, 256, 0, stream>>>(
      peaks, plen, w_mz1, b_mz1, w_mz2, b_mz2, w_int1, b_int1, w_int2, b_int2, xbf);

  pack_whh_kernel<<<(2 * 16 * 128 * 32 * 16 + 255) / 256, 256, 0, stream>>>(w_hh, whh_p);
  pack_wih_kernel<<<(2 * 128 * 2 * 32 * 16 + 255) / 256, 256, 0, stream>>>(w_ih, wih_p);
  pack_wfuse_kernel<<<(32 * 32 * 32 * 16 + 255) / 256, 256, 0, stream>>>(w_fuse, wf_p);
  init_state_kernel<<<(2 * BB * HH + 255) / 256, 256, 0, stream>>>(h0, c0, h_state, c_state);

  for (int t = 0; t < TT; ++t) {
    lstm_step_kernel<<<dim3(4, 2), 512, step_smem, stream>>>(
        xbf, wih_p, whh_p, lni_s, lni_b, lnh_s, lnh_b, lnc_s, lnc_b, h_state, c_state, t);
  }

  feat_convert_kernel<<<(BB * 2 * HH + 255) / 256, 256, 0, stream>>>(c_state, featbf);
  fuse_kernel<<<128, 32, 0, stream>>>(featbf, wf_p, b_fuse, out);
}
